// ScaledDotProductAttention_31233002177051
// MI455X (gfx1250) — compile-verified, matrix operand order checked by reading, not run on hardware
//
#include <hip/hip_runtime.h>
#include <hip/hip_bf16.h>

typedef __attribute__((ext_vector_type(16))) _Float16 v16h;
typedef __attribute__((ext_vector_type(8)))  _Float16 v8h;
typedef __attribute__((ext_vector_type(8)))  float    v8f;
typedef _Float16 half_t;

#define B_   4
#define T_   4096
#define C_   1024
#define HD_  64
#define BT_  (B_ * T_)

// -------- WMMA wrapper: D = A(16x32 f16) * B(32x16 f16) + C(16x16 f32) -----
__device__ __forceinline__ v8f wmma_f32_f16(v16h a, v16h b, v8f c) {
    return __builtin_amdgcn_wmma_f32_16x16x32_f16(
        /*neg_a=*/false, a, /*neg_b=*/false, b,
        /*c_mod=*/(short)0, c, /*reuse_a=*/false, /*reuse_b=*/false);
}

// Build a 16-half fragment from two contiguous groups of 8 halves.
// A-frag layout (16-bit, 16x32): lane L<16 -> row L, K = {0..7, 16..23};
// lane L>=16 -> row L-16, K = {8..15, 24..31}. B-frag symmetric (row->col).
__device__ __forceinline__ v16h frag16(const half_t* p0, const half_t* p1) {
    v8h a = *(const v8h*)p0;
    v8h b = *(const v8h*)p1;
    v16h r;
#pragma unroll
    for (int i = 0; i < 8; i++) { r[i] = a[i]; r[i + 8] = b[i]; }
    return r;
}

// Async 16-byte global -> LDS copy (per lane), tracked by ASYNCcnt.
// lds_off: byte offset in LDS (low 32 bits of a generic __shared__ pointer).
__device__ __forceinline__ void async_copy16(unsigned int lds_off, const void* gptr) {
    asm volatile("global_load_async_to_lds_b128 %0, %1, off"
                 :: "v"(lds_off), "v"(gptr) : "memory");
}

// ---------------------------------------------------------------------------
// Kernel 0: transpose + f16-convert weights.  Wt[m][h][c] = W_m[c*HD + h]
// ---------------------------------------------------------------------------
__global__ void wt_kernel(const float* __restrict__ Wq,
                          const float* __restrict__ Wk,
                          const float* __restrict__ Wv,
                          half_t* __restrict__ Wt) {
    int idx = blockIdx.x * blockDim.x + threadIdx.x;   // 0 .. 3*64*1024-1
    int m = idx >> 16;
    int r = idx & 65535;
    int h = r >> 10;
    int c = r & 1023;
    const float* W = (m == 0) ? Wq : ((m == 1) ? Wk : Wv);
    Wt[(size_t)m * 65536 + (size_t)h * 1024 + c] = (half_t)W[c * HD_ + h];
}

// ---------------------------------------------------------------------------
// Kernel 1: QKV projection.  Block = 128 thr (4 waves) -> 16 rows of x.
// Q, K row-major f16 [BT][64]; V transposed f16 [B][64][T].
// ---------------------------------------------------------------------------
__global__ void qkv_kernel(const float* __restrict__ x,
                           const half_t* __restrict__ Wt,
                           half_t* __restrict__ Q,
                           half_t* __restrict__ K,
                           half_t* __restrict__ Vt) {
    const int wid  = threadIdx.x >> 5;      // 0..3 -> n-tile
    const int lane = threadIdx.x & 31;
    const int ln   = lane & 15;
    const int hi   = lane >> 4;
    const int lo   = hi * 8;
    const int h0   = wid * 16;
    const size_t bt0 = (size_t)blockIdx.x * 16;

    const float*  xrow = x + (bt0 + ln) * C_;
    const half_t* wq = Wt + (size_t)0 * 65536 + (size_t)(h0 + ln) * 1024;
    const half_t* wk = Wt + (size_t)1 * 65536 + (size_t)(h0 + ln) * 1024;
    const half_t* wv = Wt + (size_t)2 * 65536 + (size_t)(h0 + ln) * 1024;

    v8f cq = {}, ck = {}, cv = {};
    for (int kc = 0; kc < C_; kc += 32) {
        const float* xp = xrow + kc + lo;
        v16h a;
#pragma unroll
        for (int i = 0; i < 8; i++) {
            a[i]     = (half_t)xp[i];
            a[i + 8] = (half_t)xp[i + 16];
        }
        v16h bq = frag16(wq + kc + lo, wq + kc + lo + 16);
        v16h bk = frag16(wk + kc + lo, wk + kc + lo + 16);
        v16h bv = frag16(wv + kc + lo, wv + kc + lo + 16);
        cq = wmma_f32_f16(a, bq, cq);
        ck = wmma_f32_f16(a, bk, ck);
        cv = wmma_f32_f16(a, bv, cv);
    }

    const int b  = (int)(bt0 / T_);
    const int t0 = (int)(bt0 % T_);
#pragma unroll
    for (int r = 0; r < 8; r++) {
        size_t row = bt0 + r + 8 * hi;
        Q[row * HD_ + h0 + ln] = (half_t)cq[r];
        K[row * HD_ + h0 + ln] = (half_t)ck[r];
    }
    v8h vh;
#pragma unroll
    for (int r = 0; r < 8; r++) vh[r] = (half_t)cv[r];
    *(v8h*)(Vt + ((size_t)b * HD_ + h0 + ln) * T_ + t0 + 8 * hi) = vh;
}

// ---------------------------------------------------------------------------
// Kernel 2: flash attention.  Block = 256 thr (8 waves), 8 query tiles.
// K/V chunks staged ONCE per block into double-buffered LDS via
// global_load_async_to_lds_b128 (ASYNCcnt), overlapping the previous chunk's
// 8 WMMAs + softmax.  Per-wave LDS transpose for the P fragment.
// ---------------------------------------------------------------------------
__global__ void attn_kernel(const half_t* __restrict__ Q,
                            const half_t* __restrict__ K,
                            const half_t* __restrict__ Vt,
                            float* __restrict__ out) {
    __shared__ __align__(16) half_t Kst[2][32][64];   // [buf][key][dim]  8 KB
    __shared__ __align__(16) half_t Vst[2][64][32];   // [buf][dim][key]  8 KB
    __shared__ __align__(16) half_t Pst[8][16][32];   // per-wave P stage 8 KB

    const int tid  = threadIdx.x;
    const int wid  = tid >> 5;
    const int lane = tid & 31;
    const int ln   = lane & 15;
    const int hi   = lane >> 4;
    const int lo   = hi * 8;

    const int g  = blockIdx.x * 8 + wid;   // global query-tile id
    const int b  = g >> 8;                 // 256 tiles per batch (8 | 256)
    const int q0 = (g & 255) * 16;
    const size_t rowbase = (size_t)b * T_ + q0;

    const half_t* Kb  = K  + (size_t)b * T_ * HD_;
    const half_t* Vtb = Vt + (size_t)b * HD_ * T_;

    // Q A-fragments (K-dim 64 = two chunks of 32)
    const half_t* qp = Q + (rowbase + ln) * HD_;
    const v16h aq0 = frag16(qp + lo,      qp + lo + 16);
    const v16h aq1 = frag16(qp + 32 + lo, qp + 32 + lo + 16);

    float mrow[8], lrow[8];
    v8f o[4] = {};
#pragma unroll
    for (int r = 0; r < 8; r++) { mrow[r] = -INFINITY; lrow[r] = 0.f; }

    const int nch_own = (q0 + 47) >> 5;
    const int q0_max  = ((blockIdx.x * 8 + 7) & 255) * 16;   // block-uniform
    const int nch_max = (q0_max + 47) >> 5;

    // ---- stage chunk jc into buffer bb: K 4KB + V 4KB, 2 async/thread ----
    auto stage = [&](int bb, int jc) {
        const int j = jc * 32;
        unsigned int kdst = (unsigned int)(uintptr_t)&Kst[bb][0][0] + tid * 16;
        const char*  ksrc = (const char*)(Kb + (size_t)j * HD_) + tid * 16;
        async_copy16(kdst, ksrc);
        const int dim = tid >> 2, part = tid & 3;
        unsigned int vdst = (unsigned int)(uintptr_t)&Vst[bb][dim][part * 8];
        const char*  vsrc = (const char*)(Vtb + (size_t)dim * T_ + j + part * 8);
        async_copy16(vdst, vsrc);
        __builtin_prefetch((const void*)(ksrc + 32 * HD_ * sizeof(half_t)), 0, 1);
    };

    stage(0, 0);
    for (int jc = 0; jc < nch_max; ++jc) {
        const int bb = jc & 1;
        // publish buffer bb: my async copies done, then block-wide barrier
        // (barrier also guarantees everyone finished reading buffer bb^1)
        asm volatile("s_wait_asynccnt 0x0" ::: "memory");
        __syncthreads();
        if (jc + 1 < nch_max) stage(bb ^ 1, jc + 1);   // overlap with compute

        if (jc < nch_own) {                            // wave-uniform branch
            const int j = jc * 32;
            // ---- S = Q * K^T  (2 key n-tiles x 2 k-chunks = 4 WMMA) ----
            const half_t* k0p = &Kst[bb][ln][0];
            const half_t* k1p = &Kst[bb][16 + ln][0];
            v16h bk00 = frag16(k0p + lo,      k0p + lo + 16);
            v16h bk01 = frag16(k0p + 32 + lo, k0p + 32 + lo + 16);
            v16h bk10 = frag16(k1p + lo,      k1p + lo + 16);
            v16h bk11 = frag16(k1p + 32 + lo, k1p + 32 + lo + 16);
            v8f s0 = {}, s1 = {};
            s0 = wmma_f32_f16(aq0, bk00, s0);
            s0 = wmma_f32_f16(aq1, bk01, s0);
            s1 = wmma_f32_f16(aq0, bk10, s1);
            s1 = wmma_f32_f16(aq1, bk11, s1);

            // ---- scale + branchless causal mask (EXEC stays all-ones) ----
#pragma unroll
            for (int r = 0; r < 8; r++) {
                const int qrow = q0 + r + 8 * hi;
                float v0 = s0[r] * 0.125f;          // HD^-0.5
                float v1 = s1[r] * 0.125f;
                s0[r] = (j + ln      > qrow) ? -INFINITY : v0;
                s1[r] = (j + 16 + ln > qrow) ? -INFINITY : v1;
            }

            // ---- online softmax ----
            float alpha[8];
#pragma unroll
            for (int r = 0; r < 8; r++) {
                float v = fmaxf(s0[r], s1[r]);
                v = fmaxf(v, __shfl_xor(v, 1));
                v = fmaxf(v, __shfl_xor(v, 2));
                v = fmaxf(v, __shfl_xor(v, 4));
                v = fmaxf(v, __shfl_xor(v, 8));
                const float mnew = fmaxf(mrow[r], v);
                alpha[r] = __expf(mrow[r] - mnew);
                mrow[r]  = mnew;
            }
#pragma unroll
            for (int r = 0; r < 8; r++) {
                const float p0 = __expf(s0[r] - mrow[r]);
                const float p1 = __expf(s1[r] - mrow[r]);
                float rs = p0 + p1;
                rs += __shfl_xor(rs, 1);
                rs += __shfl_xor(rs, 2);
                rs += __shfl_xor(rs, 4);
                rs += __shfl_xor(rs, 8);
                lrow[r] = alpha[r] * lrow[r] + rs;
                const int m = r + 8 * hi;
                Pst[wid][m][ln]      = (half_t)p0;   // C-layout -> row-major
                Pst[wid][m][16 + ln] = (half_t)p1;
            }
            __builtin_amdgcn_wave_barrier();
            asm volatile("s_wait_dscnt 0x0" ::: "memory");  // wave-local fence

            // ---- reload P as A-fragment, rescale O, O += P * V ----
            v16h ap = frag16(&Pst[wid][ln][lo], &Pst[wid][ln][lo + 16]);
#pragma unroll
            for (int nt = 0; nt < 4; nt++) {
#pragma unroll
                for (int r = 0; r < 8; r++) o[nt][r] *= alpha[r];
                const half_t* vp = &Vst[bb][nt * 16 + ln][0];
                v16h bv = frag16(vp + lo, vp + lo + 16);
                o[nt] = wmma_f32_f16(ap, bv, o[nt]);
            }
        }
    }

    // ---- finalize: O / l, write fp32 output [B][T][64] ----
#pragma unroll
    for (int r = 0; r < 8; r++) {
        const float inv = 1.0f / lrow[r];
        const size_t row = rowbase + r + 8 * hi;
#pragma unroll
        for (int nt = 0; nt < 4; nt++)
            out[row * HD_ + nt * 16 + ln] = o[nt][r] * inv;
    }
}

// ---------------------------------------------------------------------------
extern "C" void kernel_launch(void* const* d_in, const int* in_sizes, int n_in,
                              void* d_out, int out_size, void* d_ws, size_t ws_size,
                              hipStream_t stream) {
    const float* x  = (const float*)d_in[0];
    const float* Wq = (const float*)d_in[1];
    const float* Wk = (const float*)d_in[2];
    const float* Wv = (const float*)d_in[3];
    float* out = (float*)d_out;

    half_t* ws = (half_t*)d_ws;
    half_t* Wt = ws;                               // 3 * 64 * 1024 halves
    half_t* Q  = Wt + (size_t)3 * 65536;           // BT * 64
    half_t* K  = Q  + (size_t)BT_ * HD_;           // BT * 64
    half_t* Vt = K  + (size_t)BT_ * HD_;           // B * 64 * T

    wt_kernel <<< (3 * 65536) / 256, 256, 0, stream >>> (Wq, Wk, Wv, Wt);
    qkv_kernel<<< BT_ / 16,          128, 0, stream >>> (x, Wt, Q, K, Vt);
    attn_kernel<<< BT_ / 128,        256, 0, stream >>> (Q, K, Vt, out);
}
